// EncoderBlock_15779709846082
// MI455X (gfx1250) — matrix operand order repaired and checked by reading, not compiled, not run on hardware
//
#include <hip/hip_runtime.h>
#include <math.h>

// ---------------- CDNA5 types ----------------
typedef __attribute__((ext_vector_type(16))) __bf16        v16bf;
typedef __attribute__((ext_vector_type(8)))  float         v8f;
typedef __attribute__((ext_vector_type(4)))  unsigned int  u32x4;

union Frag { v16bf v; u32x4 q[2]; };

#define D_MODEL  768
#define NUM_HEADS 12
#define HEAD_DIM  64
#define MLP_DIM  3072
#define BATCH     8
#define SEQ      1024
#define ROWS     (BATCH * SEQ)   // 8192

// Async global->LDS DMA (CDNA5). Guarded: falls back to sync copies if the
// toolchain doesn't declare the builtins. Probe result (round 2): param 0 is
// "vector_size(16) int" pointer in the global (__device__/AS1) space, param 1
// the LDS (AS3) counterpart, then two imm ints (offset, cpol).
#if defined(__HIP_DEVICE_COMPILE__) &&                                        \
    __has_builtin(__builtin_amdgcn_global_load_async_to_lds_b128) &&          \
    __has_builtin(__builtin_amdgcn_s_wait_asynccnt)
#define HAVE_ASYNC_LDS 1
typedef int v4i_vs __attribute__((vector_size(16)));
typedef __attribute__((address_space(1))) v4i_vs gv4i;
typedef __attribute__((address_space(3))) v4i_vs lv4i;
#else
#define HAVE_ASYNC_LDS 0
#endif

__device__ __forceinline__ unsigned short f2bf(float f) {
    unsigned int u = __float_as_uint(f);
    unsigned int r = u + 0x7FFFu + ((u >> 16) & 1u);   // round-to-nearest-even
    return (unsigned short)(r >> 16);
}
__device__ __forceinline__ float bf2f(unsigned short h) {
    return __uint_as_float(((unsigned int)h) << 16);
}

// =====================================================================
// Weight prep: W_f32[K,N] -> Wt_bf16[N,K]  (convert + transpose, once per
// launch; whole weight set ~14MB bf16, L2-resident on MI455X's 192MB L2)
// =====================================================================
__global__ __launch_bounds__(256) void wconv_kernel(
    const float* __restrict__ W, unsigned short* __restrict__ Wt, int K, int N)
{
    size_t idx = (size_t)blockIdx.x * 256 + threadIdx.x;
    if (idx >= (size_t)K * N) return;
    int n = (int)(idx % N);
    int k = (int)(idx / N);
    Wt[(size_t)n * K + k] = f2bf(W[idx]);   // coalesced read, strided b16 write
}

// =====================================================================
// LayerNorm: f32 [row,768] -> bf16 [row,768].  One block (256 thr) / row.
// =====================================================================
__global__ __launch_bounds__(256) void ln_bf16_kernel(
    const float* __restrict__ x, const float* __restrict__ g,
    const float* __restrict__ b, unsigned short* __restrict__ out)
{
    __shared__ float red[256];
    const int row = blockIdx.x;
    const int tid = threadIdx.x;
    const float* xr = x + (size_t)row * D_MODEL;
    float v0 = xr[tid], v1 = xr[tid + 256], v2 = xr[tid + 512];
    red[tid] = v0 + v1 + v2;
    __syncthreads();
    for (int off = 128; off > 0; off >>= 1) {
        if (tid < off) red[tid] += red[tid + off];
        __syncthreads();
    }
    float mu = red[0] * (1.0f / D_MODEL);
    __syncthreads();
    float d0 = v0 - mu, d1 = v1 - mu, d2 = v2 - mu;
    red[tid] = d0 * d0 + d1 * d1 + d2 * d2;
    __syncthreads();
    for (int off = 128; off > 0; off >>= 1) {
        if (tid < off) red[tid] += red[tid + off];
        __syncthreads();
    }
    float rstd = rsqrtf(red[0] * (1.0f / D_MODEL) + 1e-5f);
    unsigned short* orow = out + (size_t)row * D_MODEL;
    orow[tid]       = f2bf(d0 * rstd * g[tid]       + b[tid]);
    orow[tid + 256] = f2bf(d1 * rstd * g[tid + 256] + b[tid + 256]);
    orow[tid + 512] = f2bf(d2 * rstd * g[tid + 512] + b[tid + 512]);
}

// =====================================================================
// GEMM: out[M,N] = epi(A_bf16[M,K] @ Wt_bf16[N,K]^T + bias) (+residual)
// 128x128x32 tiles, double-buffered LDS, async global->LDS DMA when
// available; 8 waves, wave = 32x64 => 2x4 v_wmma_f32_16x16x32_bf16.
// =====================================================================
#define BMT 128
#define BNT 128
#define BKT 32
#define LDT 40      // LDS row stride (elements); 80B rows keep 16B alignment
#define TILE_US (BMT * LDT)

// Stage one 128x32 bf16 tile (rows from g, row-major, stride row_stride).
__device__ __forceinline__ void stage_tile(
    const unsigned short* __restrict__ g, size_t row_stride,
    int row0, int k0, unsigned short* lds, int tid)
{
    #pragma unroll
    for (int j = 0; j < 2; ++j) {
        int chunk = tid + j * 256;          // 512 x 8-element chunks
        int r  = chunk >> 2;
        int c8 = (chunk & 3) * 8;
        const unsigned short* src = g + (size_t)(row0 + r) * row_stride + k0 + c8;
        unsigned short* dst = &lds[r * LDT + c8];
#if HAVE_ASYNC_LDS
        __builtin_amdgcn_global_load_async_to_lds_b128((gv4i*)src, (lv4i*)dst, 0, 0);
#else
        *(u32x4*)dst = *(const u32x4*)src;
#endif
    }
}

__global__ __launch_bounds__(256) void gemm_bf16_kernel(
    const unsigned short* __restrict__ A,    // [M,K] bf16
    const unsigned short* __restrict__ Wt,   // [N,K] bf16 (pre-transposed)
    const float* __restrict__ bias,          // [N]
    const float* __restrict__ residual,      // [M,N] f32 or null
    void* __restrict__ out,
    int N, int K, int out_bf16, int gelu)
{
    __shared__ unsigned short As[2 * TILE_US];   // [buf][row][k]
    __shared__ unsigned short Bs[2 * TILE_US];   // [buf][col][k]

    const int tid  = threadIdx.x;
    const int lane = tid & 31;
    const int wave = tid >> 5;
    const int bm = blockIdx.y * BMT;
    const int bn = blockIdx.x * BNT;
    const int wrow = (wave >> 1) * 32;   // 4 row-groups
    const int wcol = (wave & 1) * 64;    // 2 col-groups

    const int r16 = lane & 15;
    const int kbA = (lane >> 4) * 8;     // A-operand chunks {0-7,16-23}/{8-15,24-31}
    const int ksB = (lane >> 4) * 8;     // B-operand chunks {0-7,16-23}/{8-15,24-31}

    v8f acc[2][4];
    #pragma unroll
    for (int mi = 0; mi < 2; ++mi)
        #pragma unroll
        for (int ni = 0; ni < 4; ++ni)
            #pragma unroll
            for (int e = 0; e < 8; ++e) acc[mi][ni][e] = 0.0f;

    const int nt = K / BKT;
    // prologue: tile 0 -> buffer 0
    stage_tile(A,  K, bm, 0, &As[0], tid);
    stage_tile(Wt, K, bn, 0, &Bs[0], tid);

    for (int t = 0; t < nt; ++t) {
        const int cur = t & 1;
        __syncthreads();   // everyone done reading buffer 1-cur
        if (t + 1 < nt) {
            stage_tile(A,  K, bm, (t + 1) * BKT, &As[(1 - cur) * TILE_US], tid);
            stage_tile(Wt, K, bn, (t + 1) * BKT, &Bs[(1 - cur) * TILE_US], tid);
#if HAVE_ASYNC_LDS
            __builtin_amdgcn_s_wait_asynccnt(4);   // tile t's 4 issues landed
#endif
        } else {
#if HAVE_ASYNC_LDS
            __builtin_amdgcn_s_wait_asynccnt(0);
#endif
        }
        __syncthreads();   // tile t visible to all waves

        const unsigned short* Ab = &As[cur * TILE_US];
        const unsigned short* Bb = &Bs[cur * TILE_US];
        Frag a[2], bf[4];
        #pragma unroll
        for (int mi = 0; mi < 2; ++mi) {
            int row = wrow + mi * 16 + r16;
            a[mi].q[0] = *(const u32x4*)(&Ab[row * LDT + kbA]);
            a[mi].q[1] = *(const u32x4*)(&Ab[row * LDT + kbA + 16]);
        }
        #pragma unroll
        for (int ni = 0; ni < 4; ++ni) {
            int col = wcol + ni * 16 + r16;
            bf[ni].q[0] = *(const u32x4*)(&Bb[col * LDT + ksB]);
            bf[ni].q[1] = *(const u32x4*)(&Bb[col * LDT + ksB + 16]);
        }
        #pragma unroll
        for (int mi = 0; mi < 2; ++mi)
            #pragma unroll
            for (int ni = 0; ni < 4; ++ni)
                acc[mi][ni] = __builtin_amdgcn_wmma_f32_16x16x32_bf16(
                    false, a[mi].v, false, bf[ni].v, (short)0, acc[mi][ni],
                    false, false);
    }

    // ---- epilogue: bias (+gelu) (+residual), store f32 or bf16 ----
    const int rowh = (lane >> 4) * 8;
    #pragma unroll
    for (int mi = 0; mi < 2; ++mi) {
        #pragma unroll
        for (int ni = 0; ni < 4; ++ni) {
            int col = bn + wcol + ni * 16 + r16;
            float bv = bias[col];
            #pragma unroll
            for (int j = 0; j < 8; ++j) {
                int row = bm + wrow + mi * 16 + j + rowh;
                float v = acc[mi][ni][j] + bv;
                if (gelu) v = 0.5f * v * (1.0f + erff(v * 0.70710678118654752f));
                if (residual) v += residual[(size_t)row * N + col];
                if (out_bf16)
                    ((unsigned short*)out)[(size_t)row * N + col] = f2bf(v);
                else
                    ((float*)out)[(size_t)row * N + col] = v;
            }
        }
    }
}

// =====================================================================
// Flash attention: block = (b, h, 128-query tile), 8 waves x 16 rows.
// =====================================================================
#define KTILE 64
#define KSTR  80    // 160B rows: 16B aligned
#define PSTR  72    // 144B rows: 16B aligned

__global__ __launch_bounds__(256) void attn_kernel(
    const unsigned short* __restrict__ q,
    const unsigned short* __restrict__ k,
    const unsigned short* __restrict__ v,
    unsigned short* __restrict__ out)
{
    __shared__ unsigned short Kl[KTILE * KSTR];        // [key][d]
    __shared__ unsigned short Vt[HEAD_DIM * KSTR];     // [d][key]
    __shared__ unsigned short Pl[8 * 16 * PSTR];       // per-wave P (16x64)

    const int tid  = threadIdx.x;
    const int lane = tid & 31;
    const int wave = tid >> 5;
    const int h = blockIdx.y;
    const int b = blockIdx.z;

    const int r16  = lane & 15;
    const int kbA  = (lane >> 4) * 8;
    const int ksB  = (lane >> 4) * 8;
    const int rowh = (lane >> 4) * 8;

    const size_t base_bh = ((size_t)b * SEQ) * D_MODEL + h * HEAD_DIM;
    const int qrow = blockIdx.x * 128 + wave * 16;

    Frag qa[2];
    #pragma unroll
    for (int c = 0; c < 2; ++c) {
        const unsigned short* qp =
            q + base_bh + (size_t)(qrow + r16) * D_MODEL + c * 32 + kbA;
        qa[c].q[0] = *(const u32x4*)(qp);
        qa[c].q[1] = *(const u32x4*)(qp + 16);
    }

    float mrow[8], lrow[8];
    v8f o[4];
    #pragma unroll
    for (int j = 0; j < 8; ++j) { mrow[j] = -1e30f; lrow[j] = 0.0f; }
    #pragma unroll
    for (int n = 0; n < 4; ++n)
        #pragma unroll
        for (int j = 0; j < 8; ++j) o[n][j] = 0.0f;

    unsigned short* Pw = &Pl[wave * 16 * PSTR];

    for (int kt = 0; kt < SEQ; kt += KTILE) {
        __syncthreads();
        {   // stage K [key][d] and V transposed [d][key]
            int key  = tid >> 2;
            int dseg = (tid & 3) * 16;
            const unsigned short* kp =
                k + base_bh + (size_t)(kt + key) * D_MODEL + dseg;
            *(u32x4*)(&Kl[key * KSTR + dseg])     = *(const u32x4*)(kp);
            *(u32x4*)(&Kl[key * KSTR + dseg + 8]) = *(const u32x4*)(kp + 8);
            const unsigned short* vp =
                v + base_bh + (size_t)(kt + key) * D_MODEL + dseg;
            #pragma unroll
            for (int i = 0; i < 16; ++i)
                Vt[(dseg + i) * KSTR + key] = vp[i];
        }
        __syncthreads();

        // scores S (16x64) = Q @ K^T
        v8f s[4];
        #pragma unroll
        for (int n = 0; n < 4; ++n) {
            #pragma unroll
            for (int j = 0; j < 8; ++j) s[n][j] = 0.0f;
            int col = n * 16 + r16;
            #pragma unroll
            for (int c = 0; c < 2; ++c) {
                Frag kf;
                const unsigned short* kp = &Kl[col * KSTR + c * 32 + ksB];
                kf.q[0] = *(const u32x4*)(kp);
                kf.q[1] = *(const u32x4*)(kp + 16);
                s[n] = __builtin_amdgcn_wmma_f32_16x16x32_bf16(
                    false, qa[c].v, false, kf.v, (short)0, s[n], false, false);
            }
        }

        // online softmax (half-wave butterflies; wave32)
        const float scale = 0.125f;
        float tmax[8];
        #pragma unroll
        for (int j = 0; j < 8; ++j) {
            float t = fmaxf(fmaxf(s[0][j], s[1][j]), fmaxf(s[2][j], s[3][j]));
            t *= scale;
            #pragma unroll
            for (int m = 8; m >= 1; m >>= 1) t = fmaxf(t, __shfl_xor(t, m, 32));
            tmax[j] = t;
        }
        float rsum[8];
        #pragma unroll
        for (int j = 0; j < 8; ++j) {
            float mnew = fmaxf(mrow[j], tmax[j]);
            float corr = __expf(mrow[j] - mnew);
            lrow[j] *= corr;
            #pragma unroll
            for (int n = 0; n < 4; ++n) o[n][j] *= corr;
            mrow[j] = mnew;
            rsum[j] = 0.0f;
        }
        #pragma unroll
        for (int n = 0; n < 4; ++n)
            #pragma unroll
            for (int j = 0; j < 8; ++j) {
                float p = __expf(s[n][j] * scale - mrow[j]);
                rsum[j] += p;
                Pw[(j + rowh) * PSTR + n * 16 + r16] = f2bf(p);
            }
        #pragma unroll
        for (int j = 0; j < 8; ++j) {
            float t = rsum[j];
            #pragma unroll
            for (int m = 8; m >= 1; m >>= 1) t += __shfl_xor(t, m, 32);
            lrow[j] += t;
        }

        // O (16x64) += P @ V
        #pragma unroll
        for (int c = 0; c < 2; ++c) {
            Frag pa;
            const unsigned short* pp = &Pw[r16 * PSTR + c * 32 + kbA];
            pa.q[0] = *(const u32x4*)(pp);
            pa.q[1] = *(const u32x4*)(pp + 16);
            #pragma unroll
            for (int n = 0; n < 4; ++n) {
                Frag vf;
                const unsigned short* vpp = &Vt[(n * 16 + r16) * KSTR + c * 32 + ksB];
                vf.q[0] = *(const u32x4*)(vpp);
                vf.q[1] = *(const u32x4*)(vpp + 16);
                o[n] = __builtin_amdgcn_wmma_f32_16x16x32_bf16(
                    false, pa.v, false, vf.v, (short)0, o[n], false, false);
            }
        }
    }

    #pragma unroll
    for (int j = 0; j < 8; ++j) {
        float inv = 1.0f / lrow[j];
        #pragma unroll
        for (int n = 0; n < 4; ++n) o[n][j] *= inv;
    }
    #pragma unroll
    for (int n = 0; n < 4; ++n)
        #pragma unroll
        for (int j = 0; j < 8; ++j) {
            int row = qrow + j + rowh;
            out[base_bh + (size_t)row * D_MODEL + n * 16 + r16] = f2bf(o[n][j]);
        }
}

// =====================================================================
// Host-side orchestration
// =====================================================================
extern "C" void kernel_launch(void* const* d_in, const int* in_sizes, int n_in,
                              void* d_out, int out_size, void* d_ws, size_t ws_size,
                              hipStream_t stream) {
    (void)in_sizes; (void)n_in; (void)out_size; (void)ws_size;
    const float* x    = (const float*)d_in[0];
    const float* ln1g = (const float*)d_in[1];
    const float* ln1b = (const float*)d_in[2];
    const float* ln2g = (const float*)d_in[3];
    const float* ln2b = (const float*)d_in[4];
    const float* Wq   = (const float*)d_in[5];
    const float* bq   = (const float*)d_in[6];
    const float* Wk   = (const float*)d_in[7];
    const float* bk   = (const float*)d_in[8];
    const float* Wv   = (const float*)d_in[9];
    const float* bv   = (const float*)d_in[10];
    const float* Wo   = (const float*)d_in[11];
    const float* bo   = (const float*)d_in[12];
    const float* W1   = (const float*)d_in[13];
    const float* b1   = (const float*)d_in[14];
    const float* W2   = (const float*)d_in[15];
    const float* b2   = (const float*)d_in[16];

    const size_t RD = (size_t)ROWS * D_MODEL;
    const size_t WD = (size_t)D_MODEL * D_MODEL;     // 589824
    const size_t WM = (size_t)D_MODEL * MLP_DIM;     // 2359296
    char* ws = (char*)d_ws;
    unsigned short* Wqt = (unsigned short*)ws; ws += WD * 2;
    unsigned short* Wkt = (unsigned short*)ws; ws += WD * 2;
    unsigned short* Wvt = (unsigned short*)ws; ws += WD * 2;
    unsigned short* Wot = (unsigned short*)ws; ws += WD * 2;
    unsigned short* W1t = (unsigned short*)ws; ws += WM * 2;
    unsigned short* W2t = (unsigned short*)ws; ws += WM * 2;
    unsigned short* h      = (unsigned short*)ws; ws += RD * 2;
    unsigned short* qb     = (unsigned short*)ws; ws += RD * 2;
    unsigned short* kbuf   = (unsigned short*)ws; ws += RD * 2;
    unsigned short* vbuf   = (unsigned short*)ws; ws += RD * 2;
    float*          x1     = (float*)ws;          ws += RD * 4;
    unsigned short* hidden = (unsigned short*)ws; ws += (size_t)ROWS * MLP_DIM * 2;
    unsigned short* attn_o = h;    // h dead after QKV projections
    unsigned short* h2     = qb;   // q dead after attention

    dim3 blk(256);
    dim3 gD(D_MODEL / BNT, ROWS / BMT);    // (6, 64)
    dim3 gM(MLP_DIM / BNT, ROWS / BMT);    // (24, 64)
    dim3 gA(SEQ / 128, NUM_HEADS, BATCH);  // (8, 12, 8)
    const int cbD = (int)((WD + 255) / 256);
    const int cbM = (int)((WM + 255) / 256);

    // one-time (per launch) weight convert+transpose: f32[K,N] -> bf16[N,K]
    wconv_kernel<<<cbD, blk, 0, stream>>>(Wq, Wqt, D_MODEL, D_MODEL);
    wconv_kernel<<<cbD, blk, 0, stream>>>(Wk, Wkt, D_MODEL, D_MODEL);
    wconv_kernel<<<cbD, blk, 0, stream>>>(Wv, Wvt, D_MODEL, D_MODEL);
    wconv_kernel<<<cbD, blk, 0, stream>>>(Wo, Wot, D_MODEL, D_MODEL);
    wconv_kernel<<<cbM, blk, 0, stream>>>(W1, W1t, D_MODEL, MLP_DIM);
    wconv_kernel<<<cbM, blk, 0, stream>>>(W2, W2t, MLP_DIM, D_MODEL);

    ln_bf16_kernel<<<ROWS, blk, 0, stream>>>(x, ln1g, ln1b, h);
    gemm_bf16_kernel<<<gD, blk, 0, stream>>>(h, Wqt, bq, nullptr, qb,   D_MODEL, D_MODEL, 1, 0);
    gemm_bf16_kernel<<<gD, blk, 0, stream>>>(h, Wkt, bk, nullptr, kbuf, D_MODEL, D_MODEL, 1, 0);
    gemm_bf16_kernel<<<gD, blk, 0, stream>>>(h, Wvt, bv, nullptr, vbuf, D_MODEL, D_MODEL, 1, 0);
    attn_kernel<<<gA, blk, 0, stream>>>(qb, kbuf, vbuf, attn_o);
    gemm_bf16_kernel<<<gD, blk, 0, stream>>>(attn_o, Wot, bo, x, x1, D_MODEL, D_MODEL, 0, 0);
    ln_bf16_kernel<<<ROWS, blk, 0, stream>>>(x1, ln2g, ln2b, h2);
    gemm_bf16_kernel<<<gM, blk, 0, stream>>>(h2, W1t, b1, nullptr, hidden, MLP_DIM, D_MODEL, 1, 1);
    gemm_bf16_kernel<<<gD, blk, 0, stream>>>(hidden, W2t, b2, x1, (float*)d_out, D_MODEL, MLP_DIM, 0, 0);
}
